// FastMoEHLMBlock_60318520705522
// MI455X (gfx1250) — compile-verified, matrix operand order checked by reading, not run on hardware
//
#include <hip/hip_runtime.h>
#include <hip/hip_bf16.h>

// ---------------------------------------------------------------------------
// Problem constants (from reference)
// ---------------------------------------------------------------------------
#define Bc     2
#define Tc     2048
#define Dc     1024
#define Hc     16
#define DHc    64
#define Ec     16
#define Kc     2
#define Fc     2048
#define CAPc   640
#define DPc    256
#define NPRIMc 512
#define Nc     (Bc * Tc)          // 4096 tokens
#define EPSc   1e-6f

typedef __attribute__((ext_vector_type(16))) __bf16 v16bf;
typedef __attribute__((ext_vector_type(8)))  __bf16 v8bf;
typedef __attribute__((ext_vector_type(8)))  float  v8f;
typedef __attribute__((ext_vector_type(4)))  unsigned int v4u;
typedef __attribute__((ext_vector_type(8)))  int v8i;
typedef __attribute__((ext_vector_type(4)))  int v4i;

#ifndef __has_builtin
#define __has_builtin(x) 0
#endif

#if __has_builtin(__builtin_amdgcn_tensor_load_to_lds) && \
    __has_builtin(__builtin_amdgcn_s_wait_tensorcnt)
#define HAVE_TDM 1
#else
#define HAVE_TDM 0
#endif

// ---------------------------------------------------------------------------
// Helpers
// ---------------------------------------------------------------------------
__device__ __forceinline__ __bf16 f2bf(float f) { return (__bf16)f; }
__device__ __forceinline__ float  bf2f(__bf16 b) { return (float)b; }

// Async global->LDS copy (ASYNCcnt-tracked), GV addressing: per-lane LDS byte
// address VGPR + 64-bit global address VGPR pair.  16 bytes per lane.
__device__ __forceinline__ void async_lds_b128(unsigned lds_addr,
                                               unsigned long long gaddr) {
  asm volatile("global_load_async_to_lds_b128 %0, %1, off"
               :: "v"(lds_addr), "v"(gaddr) : "memory");
}

__device__ __forceinline__ void wait_async0() {
#if __has_builtin(__builtin_amdgcn_s_wait_asynccnt)
  __builtin_amdgcn_s_wait_asynccnt(0);
#else
  asm volatile("s_wait_asynccnt 0" ::: "memory");
#endif
}

// K index within a 16-bit 16x32 A (or 32x16 B) fragment for element e of the
// packed v16bf, per CDNA5 ISA 7.12.2 (lanes 0-15: K pairs 0..7 then 16..23;
// lanes 16-31: K pairs 8..15 then 24..31).
__device__ __forceinline__ int frag_k(int e, int lane) {
  int j = e >> 1;
  return ((lane & 16) ? 8 : 0) + ((j & 4) ? 16 : 0) + ((j & 3) << 1) + (e & 1);
}

__device__ __forceinline__ v8f wmma_bf16(v16bf a, v16bf b, v8f c) {
  return __builtin_amdgcn_wmma_f32_16x16x32_bf16(false, a, false, b, (short)0,
                                                 c, false, false);
}

// ---------------------------------------------------------------------------
// fp32 -> bf16 convert (plain)
// ---------------------------------------------------------------------------
__global__ void convert_bf16_kernel(const float* __restrict__ in,
                                    __bf16* __restrict__ out, long n) {
  long i = (long)blockIdx.x * blockDim.x + threadIdx.x;
  if (i < n) out[i] = f2bf(in[i]);
}

// fp32 [Z,R,C] -> bf16 [Z,C,R] transpose-convert
__global__ void convert_t_bf16_kernel(const float* __restrict__ in,
                                      __bf16* __restrict__ out, int R, int C) {
  int z = blockIdx.z;
  long base = (long)z * R * C;
  long i = (long)blockIdx.x * blockDim.x + threadIdx.x;
  if (i < (long)R * C) {
    int r = (int)(i / C), c = (int)(i % C);
    out[base + (long)c * R + r] = f2bf(in[base + i]);
  }
}

// ---------------------------------------------------------------------------
// RMSNorm: fp32 [rows,1024] * w -> bf16
// ---------------------------------------------------------------------------
__global__ void rmsnorm_kernel(const float* __restrict__ x,
                               const float* __restrict__ w,
                               __bf16* __restrict__ out) {
  int row = blockIdx.x;
  const float* xr = x + (long)row * Dc;
  float s = 0.f;
  for (int i = threadIdx.x; i < Dc; i += 256) { float v = xr[i]; s += v * v; }
  __shared__ float red[8];
  for (int o = 16; o; o >>= 1) s += __shfl_xor(s, o, 32);
  if ((threadIdx.x & 31) == 0) red[threadIdx.x >> 5] = s;
  __syncthreads();
  if (threadIdx.x == 0) {
    float t = 0.f;
    for (int i = 0; i < 8; ++i) t += red[i];
    red[0] = t;
  }
  __syncthreads();
  float inv = rsqrtf(red[0] / (float)Dc + EPSc);
  for (int i = threadIdx.x; i < Dc; i += 256)
    out[(long)row * Dc + i] = f2bf(xr[i] * inv * w[i]);
}

// ---------------------------------------------------------------------------
// Generic WMMA GEMM: C[M,Nn] = A[M,Kd] * Bw[Nn,Kd]^T  (row-major, bf16->f32)
// 64x64 tile per block, 128 threads = 4 waves, wave w owns rows [16w,16w+16).
// Tiles are staged into LDS with ASYNCcnt-tracked async copies so the copy
// engine runs decoupled from the wave (no VGPR bounce).
// Strides sA/sB/sC select the batch slice for blockIdx.z.
// ---------------------------------------------------------------------------
__global__ __launch_bounds__(128)
void gemm_bf16_kernel(const __bf16* __restrict__ A,
                      const __bf16* __restrict__ Bw,
                      float* __restrict__ C,
                      int M, int Nn, int Kd,
                      long sA, long sB, long sC) {
  __shared__ __bf16 As[64][32];
  __shared__ __bf16 Bs[64][32];
  int z = blockIdx.z;
  A  += (long)z * sA;
  Bw += (long)z * sB;
  C  += (long)z * sC;
  int tile_n = blockIdx.x * 64;
  int tile_m = blockIdx.y * 64;
  int tid  = threadIdx.x;
  int lane = tid & 31;
  int wave = tid >> 5;

  v8f acc[4];
  for (int i = 0; i < 4; ++i) acc[i] = (v8f){};

  int lr = tid >> 1;            // row within tile for cooperative loads
  int lc = (tid & 1) * 16;      // col within 32-wide k-slab
  unsigned la = (unsigned)(unsigned long long)&As[lr][lc];
  unsigned lb = (unsigned)(unsigned long long)&Bs[lr][lc];

  for (int k0 = 0; k0 < Kd; k0 += 32) {
    const __bf16* ag = A  + (long)(tile_m + lr) * Kd + k0 + lc;
    const __bf16* bg = Bw + (long)(tile_n + lr) * Kd + k0 + lc;
    if (k0 + 32 < Kd) {               // hint next slab into cache
      __builtin_prefetch(ag + 32, 0, 1);
      __builtin_prefetch(bg + 32, 0, 1);
    }
    async_lds_b128(la,      (unsigned long long)ag);
    async_lds_b128(la + 16, (unsigned long long)(ag + 8));
    async_lds_b128(lb,      (unsigned long long)bg);
    async_lds_b128(lb + 16, (unsigned long long)(bg + 8));
    wait_async0();
    __syncthreads();

    v16bf af;
    int m = wave * 16 + (lane & 15);
#pragma unroll
    for (int e = 0; e < 16; ++e) af[e] = As[m][frag_k(e, lane)];

#pragma unroll
    for (int nt = 0; nt < 4; ++nt) {
      v16bf bfv;
      int n = nt * 16 + (lane & 15);
#pragma unroll
      for (int e = 0; e < 16; ++e) bfv[e] = Bs[n][frag_k(e, lane)];
      acc[nt] = wmma_bf16(af, bfv, acc[nt]);
    }
    __syncthreads();
  }

#pragma unroll
  for (int nt = 0; nt < 4; ++nt) {
#pragma unroll
    for (int v = 0; v < 8; ++v) {
      int r = tile_m + wave * 16 + v + ((lane & 16) ? 8 : 0);
      int c = tile_n + nt * 16 + (lane & 15);
      if (r < M && c < Nn) C[(long)r * Nn + c] = acc[nt][v];
    }
  }
}

// ---------------------------------------------------------------------------
// RoPE: qkv fp32 [N, 3*D] -> qh/kh (rotated) and vh, bf16 [B,H,T,DH]
// ---------------------------------------------------------------------------
__global__ void rope_kernel(const float* __restrict__ qkv,
                            __bf16* __restrict__ qh,
                            __bf16* __restrict__ kh,
                            __bf16* __restrict__ vh) {
  int nh = blockIdx.x;
  int h = nh % Hc, n = nh / Hc;
  int b = n / Tc, t = n % Tc;
  int d = threadIdx.x;                       // 0..63
  long base = (long)n * (3 * Dc) + h * DHc;
  float q = qkv[base + d];
  float k = qkv[base + Dc + d];
  float v = qkv[base + 2 * Dc + d];
  int dm = d & 31;
  float inv_freq = powf(10000.f, -(float)dm / 32.f);
  float ang = (float)t * inv_freq;
  float c = cosf(ang), s = sinf(ang);
  int dp = (d < 32) ? d + 32 : d - 32;
  float q2 = qkv[base + dp];
  float k2 = qkv[base + Dc + dp];
  float qr = (d < 32) ? (q * c - q2 * s) : (q * c + q2 * s);
  float kr = (d < 32) ? (k * c - k2 * s) : (k * c + k2 * s);
  long o = (((long)(b * Hc + h) * Tc) + t) * DHc + d;
  qh[o] = f2bf(qr);
  kh[o] = f2bf(kr);
  vh[o] = f2bf(v);
}

// ---------------------------------------------------------------------------
// Streaming causal attention, one wave per (b, h, 16-query tile).
// Q tile is fetched by the Tensor Data Mover (TENSORcnt-tracked) when the
// toolchain exposes the TDM builtin; K/V stream through LDS.
// S = Q K^T / 8 via WMMA, online softmax, O = P V via WMMA.
// ---------------------------------------------------------------------------
__global__ __launch_bounds__(32)
void flash_attn_kernel(const __bf16* __restrict__ qh,
                       const __bf16* __restrict__ kh,
                       const __bf16* __restrict__ vh,
                       __bf16* __restrict__ attnb) {
  __shared__ __bf16 Qs[16][64];
  __shared__ __bf16 Ks[32][64];
  __shared__ __bf16 Vs[32][64];
  __shared__ __bf16 Ps[16][32];

  int qt = blockIdx.x, h = blockIdx.y, b = blockIdx.z;
  int lane = threadIdx.x;

  const __bf16* Qp = qh + (((long)(b * Hc + h) * Tc) + qt * 16) * DHc;
  const __bf16* Kp = kh + ((long)(b * Hc + h) * Tc) * DHc;
  const __bf16* Vp = vh + ((long)(b * Hc + h) * Tc) * DHc;

#if HAVE_TDM
  {
    // TDM D#: 1D tile of 1024 bf16 elements (2 KB), global -> LDS.
    unsigned lq = (unsigned)(unsigned long long)&Qs[0][0];
    unsigned long long ga = (unsigned long long)Qp;
    v4u g0;
    g0[0] = 1u;                                       // count=1 (user mode)
    g0[1] = lq;                                       // lds_addr
    g0[2] = (unsigned)(ga & 0xFFFFFFFFu);             // global_addr[31:0]
    g0[3] = (unsigned)((ga >> 32) & 0x01FFFFFFull) |  // global_addr[56:32]
            (2u << 30);                               // type=2 (image)
    v8i g1;
    g1[0] = (int)(1u << 16);        // wg_mask=0, data_size=1 (2 bytes)
    g1[1] = (int)(1024u << 16);     // tensor_dim0[15:0]=1024
    g1[2] = (int)(1u << 16);        // tensor_dim0[31:16]=0, tensor_dim1=1
    g1[3] = (int)(1024u << 16);     // tile_dim0=1024
    g1[4] = 0;                      // tile_dim1=0 (unused), tile_dim2=0
    g1[5] = 1024;                   // tensor_dim0_stride[31:0]
    g1[6] = 0;
    g1[7] = 0;
    v4i g2 = {0, 0, 0, 0};
    v4i g3 = {0, 0, 0, 0};
#if defined(__clang_major__) && (__clang_major__ >= 23)
    v8i g4 = {0, 0, 0, 0, 0, 0, 0, 0};
    __builtin_amdgcn_tensor_load_to_lds(g0, g1, g2, g3, g4, 0);
#else
    __builtin_amdgcn_tensor_load_to_lds(g0, g1, g2, g3, 0);
#endif
    __builtin_amdgcn_s_wait_tensorcnt((short)0);
  }
#else
  {
    v8bf* qv = (v8bf*)&Qs[0][0];
    const v8bf* qg = (const v8bf*)Qp;
    for (int i = lane; i < 16 * 64 / 8; i += 32) qv[i] = qg[i];
  }
#endif
  __syncthreads();

  v16bf aq[2];
#pragma unroll
  for (int c = 0; c < 2; ++c) {
    int m = lane & 15;
#pragma unroll
    for (int e = 0; e < 16; ++e) aq[c][e] = Qs[m][c * 32 + frag_k(e, lane)];
  }

  v8f o0 = (v8f){}, o1 = (v8f){}, o2 = (v8f){}, o3 = (v8f){};
  float mrow[8], lrow[8];
#pragma unroll
  for (int v = 0; v < 8; ++v) { mrow[v] = -1e30f; lrow[v] = 0.f; }

  int kmax = qt * 16 + 15;
  for (int kb = 0; kb * 32 <= kmax; ++kb) {
    {
      v8bf* kv = (v8bf*)&Ks[0][0];
      v8bf* vv = (v8bf*)&Vs[0][0];
      const v8bf* kg = (const v8bf*)Kp + (long)kb * (32 * 64 / 8);
      const v8bf* vg = (const v8bf*)Vp + (long)kb * (32 * 64 / 8);
      for (int i = lane; i < 32 * 64 / 8; i += 32) { kv[i] = kg[i]; vv[i] = vg[i]; }
    }
    __syncthreads();

    v8f s2[2];
    s2[0] = (v8f){}; s2[1] = (v8f){};
#pragma unroll
    for (int t2 = 0; t2 < 2; ++t2) {
#pragma unroll
      for (int c = 0; c < 2; ++c) {
        v16bf bk;
        int n = lane & 15;
#pragma unroll
        for (int e = 0; e < 16; ++e)
          bk[e] = Ks[t2 * 16 + n][c * 32 + frag_k(e, lane)];
        s2[t2] = wmma_bf16(aq[c], bk, s2[t2]);
      }
    }

    const float sc = 0.125f;   // 1/sqrt(64)
    int key0 = kb * 32 + (lane & 15);
    int key1 = key0 + 16;
#pragma unroll
    for (int v = 0; v < 8; ++v) {
      int row = qt * 16 + v + ((lane & 16) ? 8 : 0);
      float x0 = s2[0][v] * sc;
      float x1 = s2[1][v] * sc;
      if (key0 > row) x0 = -1e30f;
      if (key1 > row) x1 = -1e30f;
      float rmax = fmaxf(x0, x1);
      for (int o = 8; o; o >>= 1) rmax = fmaxf(rmax, __shfl_xor(rmax, o, 16));
      float mn = fmaxf(mrow[v], rmax);
      float corr = __expf(mrow[v] - mn);
      float e0 = __expf(x0 - mn);
      float e1 = __expf(x1 - mn);
      float rs = e0 + e1;
      for (int o = 8; o; o >>= 1) rs += __shfl_xor(rs, o, 16);
      lrow[v] = lrow[v] * corr + rs;
      mrow[v] = mn;
      o0[v] *= corr; o1[v] *= corr; o2[v] *= corr; o3[v] *= corr;
      s2[0][v] = e0; s2[1][v] = e1;
    }

    // Re-lay-out P (C layout) into an A-fragment via LDS.
#pragma unroll
    for (int v = 0; v < 8; ++v) {
      int r = v + ((lane & 16) ? 8 : 0);
      Ps[r][lane & 15]        = f2bf(s2[0][v]);
      Ps[r][16 + (lane & 15)] = f2bf(s2[1][v]);
    }
    __syncthreads();
    v16bf ap;
    {
      int m = lane & 15;
#pragma unroll
      for (int e = 0; e < 16; ++e) ap[e] = Ps[m][frag_k(e, lane)];
    }

#pragma unroll
    for (int nt = 0; nt < 4; ++nt) {
      v16bf bv;
      int n = lane & 15;
#pragma unroll
      for (int e = 0; e < 16; ++e) bv[e] = Vs[frag_k(e, lane)][nt * 16 + n];
      v8f* op = (nt == 0) ? &o0 : (nt == 1) ? &o1 : (nt == 2) ? &o2 : &o3;
      *op = wmma_bf16(ap, bv, *op);
    }
    __syncthreads();
  }

  // out layout: [B,T,H*DH] row n = b*T+t, col = h*64+d
#pragma unroll
  for (int nt = 0; nt < 4; ++nt) {
    v8f* op = (nt == 0) ? &o0 : (nt == 1) ? &o1 : (nt == 2) ? &o2 : &o3;
#pragma unroll
    for (int v = 0; v < 8; ++v) {
      int row = qt * 16 + v + ((lane & 16) ? 8 : 0);
      int d = nt * 16 + (lane & 15);
      long idx = ((long)b * Tc + row) * Dc + h * DHc + d;
      attnb[idx] = f2bf((*op)[v] / lrow[v]);
    }
  }
}

// ---------------------------------------------------------------------------
// residual add (fp32)
// ---------------------------------------------------------------------------
__global__ void add_kernel(const float* __restrict__ a,
                           const float* __restrict__ b,
                           float* __restrict__ c, long n) {
  long i = (long)blockIdx.x * blockDim.x + threadIdx.x;
  if (i < n) c[i] = a[i] + b[i];
}

// ---------------------------------------------------------------------------
// MoE router: logits = xn @ rw^T, softmax, top-2 (renormalized)
// block = 128 threads per token; 8 lanes per expert.
// ---------------------------------------------------------------------------
__global__ void router_kernel(const __bf16* __restrict__ xn,
                              const float* __restrict__ rw,
                              float* __restrict__ probs,
                              int* __restrict__ topi,
                              float* __restrict__ topw) {
  int n = blockIdx.x;
  int tid = threadIdx.x;
  int e = tid >> 3, sub = tid & 7;
  const __bf16* xr = xn + (long)n * Dc;
  const float* wr = rw + (long)e * Dc;
  float s = 0.f;
  for (int i = sub; i < Dc; i += 8) s += bf2f(xr[i]) * wr[i];
  for (int o = 4; o; o >>= 1) s += __shfl_xor(s, o, 8);
  __shared__ float lg[Ec];
  if (sub == 0) lg[e] = s;
  __syncthreads();
  if (tid == 0) {
    float mx = -1e30f;
    for (int i = 0; i < Ec; ++i) mx = fmaxf(mx, lg[i]);
    float p[Ec], sm = 0.f;
    for (int i = 0; i < Ec; ++i) { p[i] = __expf(lg[i] - mx); sm += p[i]; }
    for (int i = 0; i < Ec; ++i) { p[i] /= sm; probs[(long)n * Ec + i] = p[i]; }
    int i0 = 0;
    for (int i = 1; i < Ec; ++i) if (p[i] > p[i0]) i0 = i;
    int i1 = -1;
    for (int i = 0; i < Ec; ++i) {
      if (i == i0) continue;
      if (i1 < 0 || p[i] > p[i1]) i1 = i;
    }
    float t = p[i0] + p[i1];
    topi[n * 2] = i0;     topi[n * 2 + 1] = i1;
    topw[n * 2] = p[i0] / t; topw[n * 2 + 1] = p[i1] / t;
  }
}

// Deterministic capacity slot assignment (matches cumsum order).
__global__ void assign_slots_kernel(const int* __restrict__ topi,
                                    int* __restrict__ slot,
                                    int* __restrict__ keep,
                                    int* __restrict__ cnt) {
  if (blockIdx.x == 0 && threadIdx.x == 0) {
    int c[Ec];
    for (int i = 0; i < Ec; ++i) c[i] = 0;
    for (int i = 0; i < Nc * Kc; ++i) {
      int e = topi[i];
      int p = c[e]++;
      keep[i] = (p < CAPc) ? 1 : 0;
      slot[i] = (p < CAPc) ? p : (CAPc - 1);
    }
    for (int i = 0; i < Ec; ++i) cnt[i] = c[i];   // uncapped counts for aux
  }
}

// Scatter token rows into expert capacity buffer.
__global__ void scatter_kernel(const __bf16* __restrict__ xn,
                               const int* __restrict__ topi,
                               const int* __restrict__ slot,
                               const int* __restrict__ keep,
                               __bf16* __restrict__ buf) {
  int i = blockIdx.x;            // token*K index
  if (!keep[i]) return;
  int tok = i >> 1;
  const v8bf* src = (const v8bf*)(xn + (long)tok * Dc);
  v8bf* dst = (v8bf*)(buf + ((long)topi[i] * CAPc + slot[i]) * Dc);
  for (int j = threadIdx.x; j < Dc / 8; j += 128) dst[j] = src[j];
}

__global__ void silu_bf16_kernel(const float* __restrict__ in,
                                 __bf16* __restrict__ out, long n) {
  long i = (long)blockIdx.x * blockDim.x + threadIdx.x;
  if (i < n) {
    float v = in[i];
    out[i] = f2bf(v / (1.f + __expf(-v)));
  }
}

// x2 = x1 + sum_k keep*w * yb[expert, slot]
__global__ void combine_kernel(const float* __restrict__ x1,
                               const float* __restrict__ yb,
                               const int* __restrict__ topi,
                               const int* __restrict__ slot,
                               const int* __restrict__ keep,
                               const float* __restrict__ topw,
                               float* __restrict__ x2) {
  int n = blockIdx.x;
  for (int i = threadIdx.x; i < Dc; i += 256) {
    float acc = x1[(long)n * Dc + i];
    for (int k = 0; k < Kc; ++k) {
      int id = n * Kc + k;
      if (keep[id])
        acc += topw[id] * yb[((long)topi[id] * CAPc + slot[id]) * Dc + i];
    }
    x2[(long)n * Dc + i] = acc;
  }
}

// aux = E * sum_e frac[e] * pmean[e]
__global__ void aux_kernel(const float* __restrict__ probs,
                           const int* __restrict__ cnt,
                           float* __restrict__ aux_out) {
  __shared__ float wsum[8][Ec];
  float acc[Ec];
  for (int e = 0; e < Ec; ++e) acc[e] = 0.f;
  for (int n = threadIdx.x; n < Nc; n += 256)
    for (int e = 0; e < Ec; ++e) acc[e] += probs[(long)n * Ec + e];
  for (int e = 0; e < Ec; ++e)
    for (int o = 16; o; o >>= 1) acc[e] += __shfl_xor(acc[e], o, 32);
  int w = threadIdx.x >> 5;
  if ((threadIdx.x & 31) == 0)
    for (int e = 0; e < Ec; ++e) wsum[w][e] = acc[e];
  __syncthreads();
  if (threadIdx.x == 0) {
    float a = 0.f;
    for (int e = 0; e < Ec; ++e) {
      float p = 0.f;
      for (int ww = 0; ww < 8; ++ww) p += wsum[ww][e];
      a += ((float)cnt[e] / (float)Nc) * (p / (float)Nc);
    }
    aux_out[0] = (float)Ec * a;
  }
}

// ---------------------------------------------------------------------------
// tou cross-attn: softmax over 512 keys (scale 1/16), probs -> bf16
// ---------------------------------------------------------------------------
__global__ void softmax512_kernel(const float* __restrict__ sc,
                                  __bf16* __restrict__ pb) {
  int n = blockIdx.x;
  int tid = threadIdx.x;               // 256
  const float scale = 0.0625f;         // 1/sqrt(256)
  float a = sc[(long)n * NPRIMc + tid] * scale;
  float b = sc[(long)n * NPRIMc + 256 + tid] * scale;
  __shared__ float red[8];
  float m = fmaxf(a, b);
  for (int o = 16; o; o >>= 1) m = fmaxf(m, __shfl_xor(m, o, 32));
  if ((tid & 31) == 0) red[tid >> 5] = m;
  __syncthreads();
  if (tid == 0) {
    float t = red[0];
    for (int i = 1; i < 8; ++i) t = fmaxf(t, red[i]);
    red[0] = t;
  }
  __syncthreads();
  m = red[0];
  __syncthreads();
  float e0 = __expf(a - m), e1 = __expf(b - m);
  float s = e0 + e1;
  for (int o = 16; o; o >>= 1) s += __shfl_xor(s, o, 32);
  if ((tid & 31) == 0) red[tid >> 5] = s;
  __syncthreads();
  if (tid == 0) {
    float t = 0.f;
    for (int i = 0; i < 8; ++i) t += red[i];
    red[0] = t;
  }
  __syncthreads();
  s = red[0];
  pb[(long)n * NPRIMc + tid]       = f2bf(e0 / s);
  pb[(long)n * NPRIMc + 256 + tid] = f2bf(e1 / s);
}

// out = x2 + sigmoid(xn3 . gw + gb) * proj
__global__ void final_kernel(const float* __restrict__ x2,
                             const float* __restrict__ proj,
                             const __bf16* __restrict__ xn3,
                             const float* __restrict__ gw,
                             const float* __restrict__ gb,
                             float* __restrict__ out) {
  int n = blockIdx.x;
  float s = 0.f;
  for (int i = threadIdx.x; i < Dc; i += 256)
    s += bf2f(xn3[(long)n * Dc + i]) * gw[i];
  __shared__ float red[8];
  for (int o = 16; o; o >>= 1) s += __shfl_xor(s, o, 32);
  if ((threadIdx.x & 31) == 0) red[threadIdx.x >> 5] = s;
  __syncthreads();
  if (threadIdx.x == 0) {
    float t = 0.f;
    for (int i = 0; i < 8; ++i) t += red[i];
    red[0] = t;
  }
  __syncthreads();
  float gate = 1.f / (1.f + __expf(-(red[0] + gb[0])));
  for (int i = threadIdx.x; i < Dc; i += 256)
    out[(long)n * Dc + i] = x2[(long)n * Dc + i] + gate * proj[(long)n * Dc + i];
}

// ---------------------------------------------------------------------------
// Host: orchestrate the whole block on `stream`
// ---------------------------------------------------------------------------
static inline void run_gemm(const __bf16* A, const __bf16* Bw, float* C,
                            int M, int Nn, int Kd,
                            long sA, long sB, long sC, int Z,
                            hipStream_t stream) {
  dim3 g(Nn / 64, M / 64, Z);
  gemm_bf16_kernel<<<g, 128, 0, stream>>>(A, Bw, C, M, Nn, Kd, sA, sB, sC);
}

static inline void run_convert(const float* in, __bf16* out, long n,
                               hipStream_t stream) {
  convert_bf16_kernel<<<(int)((n + 255) / 256), 256, 0, stream>>>(in, out, n);
}

extern "C" void kernel_launch(void* const* d_in, const int* in_sizes, int n_in,
                              void* d_out, int out_size, void* d_ws,
                              size_t ws_size, hipStream_t stream) {
  (void)in_sizes; (void)n_in; (void)out_size; (void)ws_size;

  const float* x          = (const float*)d_in[0];
  const float* tou_embeds = (const float*)d_in[1];
  const float* norm1_w    = (const float*)d_in[2];
  const float* qkv_w      = (const float*)d_in[3];
  const float* attn_o_w   = (const float*)d_in[4];
  const float* norm2_w    = (const float*)d_in[5];
  const float* router_w   = (const float*)d_in[6];
  const float* moe_w1     = (const float*)d_in[7];
  const float* moe_w2     = (const float*)d_in[8];
  const float* norm3_w    = (const float*)d_in[9];
  const float* tou_q_w    = (const float*)d_in[10];
  const float* tou_k_w    = (const float*)d_in[11];
  const float* tou_v_w    = (const float*)d_in[12];
  const float* tou_o_w    = (const float*)d_in[13];
  const float* tou_gate_w = (const float*)d_in[14];
  const float* tou_gate_b = (const float*)d_in[15];
  float* out = (float*)d_out;

  char* ws = (char*)d_ws;
  size_t off = 0;
  auto alloc = [&](size_t bytes) -> void* {
    void* p = ws + off;
    off = (off + bytes + 255) & ~(size_t)255;
    return p;
  };

  // bf16 weight copies
  __bf16* wqkv = (__bf16*)alloc((size_t)3 * Dc * Dc * 2);
  __bf16* wo   = (__bf16*)alloc((size_t)Dc * Dc * 2);
  __bf16* w1t  = (__bf16*)alloc((size_t)Ec * Fc * Dc * 2);   // [E,F,D]
  __bf16* w2t  = (__bf16*)alloc((size_t)Ec * Dc * Fc * 2);   // [E,D,F]
  __bf16* wtq  = (__bf16*)alloc((size_t)DPc * Dc * 2);
  __bf16* wtk  = (__bf16*)alloc((size_t)DPc * DPc * 2);
  __bf16* wtv  = (__bf16*)alloc((size_t)DPc * DPc * 2);
  __bf16* wto  = (__bf16*)alloc((size_t)Dc * DPc * 2);
  __bf16* primb = (__bf16*)alloc((size_t)NPRIMc * DPc * 2);
  // activations
  __bf16* xn    = (__bf16*)alloc((size_t)Nc * Dc * 2);
  float*  qkvf  = (float*)alloc((size_t)Nc * 3 * Dc * 4);
  __bf16* qh    = (__bf16*)alloc((size_t)Nc * Dc * 2);
  __bf16* kh    = (__bf16*)alloc((size_t)Nc * Dc * 2);
  __bf16* vh    = (__bf16*)alloc((size_t)Nc * Dc * 2);
  __bf16* attnb = (__bf16*)alloc((size_t)Nc * Dc * 2);
  float*  projf = (float*)alloc((size_t)Nc * Dc * 4);
  float*  x1    = (float*)alloc((size_t)Nc * Dc * 4);
  float*  probs = (float*)alloc((size_t)Nc * Ec * 4);
  int*    topi  = (int*)alloc((size_t)Nc * Kc * 4);
  float*  topw  = (float*)alloc((size_t)Nc * Kc * 4);
  int*    slot  = (int*)alloc((size_t)Nc * Kc * 4);
  int*    keep  = (int*)alloc((size_t)Nc * Kc * 4);
  int*    cnt   = (int*)alloc((size_t)Ec * 4);
  __bf16* buf   = (__bf16*)alloc((size_t)Ec * CAPc * Dc * 2);
  float*  hf    = (float*)alloc((size_t)Ec * CAPc * Fc * 4);
  __bf16* hb    = (__bf16*)alloc((size_t)Ec * CAPc * Fc * 2);
  float*  ybf   = (float*)alloc((size_t)Ec * CAPc * Dc * 4);
  float*  x2    = (float*)alloc((size_t)Nc * Dc * 4);
  float*  qtf   = (float*)alloc((size_t)Nc * DPc * 4);
  __bf16* qtb   = (__bf16*)alloc((size_t)Nc * DPc * 2);
  float*  kf    = (float*)alloc((size_t)NPRIMc * DPc * 4);
  float*  vf    = (float*)alloc((size_t)NPRIMc * DPc * 4);
  __bf16* kb    = (__bf16*)alloc((size_t)NPRIMc * DPc * 2);
  __bf16* vtb   = (__bf16*)alloc((size_t)DPc * NPRIMc * 2);  // V^T [256,512]
  float*  scf   = (float*)alloc((size_t)Nc * NPRIMc * 4);
  __bf16* pb    = (__bf16*)alloc((size_t)Nc * NPRIMc * 2);
  float*  ctxf  = (float*)alloc((size_t)Nc * DPc * 4);
  __bf16* ctxb  = (__bf16*)alloc((size_t)Nc * DPc * 2);
  float*  tprojf = (float*)alloc((size_t)Nc * Dc * 4);

  // --- weight conversion ---
  run_convert(qkv_w, wqkv, (long)3 * Dc * Dc, stream);
  run_convert(attn_o_w, wo, (long)Dc * Dc, stream);
  run_convert(tou_q_w, wtq, (long)DPc * Dc, stream);
  run_convert(tou_k_w, wtk, (long)DPc * DPc, stream);
  run_convert(tou_v_w, wtv, (long)DPc * DPc, stream);
  run_convert(tou_o_w, wto, (long)Dc * DPc, stream);
  run_convert(tou_embeds, primb, (long)NPRIMc * DPc, stream);
  {
    long per = (long)Dc * Fc;
    dim3 g((unsigned)((per + 255) / 256), 1, Ec);
    convert_t_bf16_kernel<<<g, 256, 0, stream>>>(moe_w1, w1t, Dc, Fc);  // [D,F]->[F,D]
    convert_t_bf16_kernel<<<g, 256, 0, stream>>>(moe_w2, w2t, Fc, Dc);  // [F,D]->[D,F]
  }

  // --- stage 1: RMSNorm + QKV + RoPE + causal attention + O proj + residual
  rmsnorm_kernel<<<Nc, 256, 0, stream>>>(x, norm1_w, xn);
  run_gemm(xn, wqkv, qkvf, Nc, 3 * Dc, Dc, 0, 0, 0, 1, stream);
  rope_kernel<<<Nc * Hc, DHc, 0, stream>>>(qkvf, qh, kh, vh);
  {
    dim3 g(Tc / 16, Hc, Bc);
    flash_attn_kernel<<<g, 32, 0, stream>>>(qh, kh, vh, attnb);
  }
  run_gemm(attnb, wo, projf, Nc, Dc, Dc, 0, 0, 0, 1, stream);
  add_kernel<<<(Nc * Dc + 255) / 256, 256, 0, stream>>>(x, projf, x1,
                                                        (long)Nc * Dc);

  // --- stage 2: MoE ---
  rmsnorm_kernel<<<Nc, 256, 0, stream>>>(x1, norm2_w, xn);
  router_kernel<<<Nc, 128, 0, stream>>>(xn, router_w, probs, topi, topw);
  assign_slots_kernel<<<1, 1, 0, stream>>>(topi, slot, keep, cnt);
  hipMemsetAsync(buf, 0, (size_t)Ec * CAPc * Dc * 2, stream);
  scatter_kernel<<<Nc * Kc, 128, 0, stream>>>(xn, topi, slot, keep, buf);
  run_gemm(buf, w1t, hf, CAPc, Fc, Dc,
           (long)CAPc * Dc, (long)Fc * Dc, (long)CAPc * Fc, Ec, stream);
  silu_bf16_kernel<<<(int)(((long)Ec * CAPc * Fc + 255) / 256), 256, 0,
                     stream>>>(hf, hb, (long)Ec * CAPc * Fc);
  run_gemm(hb, w2t, ybf, CAPc, Dc, Fc,
           (long)CAPc * Fc, (long)Dc * Fc, (long)CAPc * Dc, Ec, stream);
  combine_kernel<<<Nc, 256, 0, stream>>>(x1, ybf, topi, slot, keep, topw, x2);
  aux_kernel<<<1, 256, 0, stream>>>(probs, cnt, out + (long)Nc * Dc);

  // --- stage 3: tou cross-attention ---
  rmsnorm_kernel<<<Nc, 256, 0, stream>>>(x2, norm3_w, xn);
  run_gemm(xn, wtq, qtf, Nc, DPc, Dc, 0, 0, 0, 1, stream);
  run_convert(qtf, qtb, (long)Nc * DPc, stream);
  run_gemm(primb, wtk, kf, NPRIMc, DPc, DPc, 0, 0, 0, 1, stream);
  run_gemm(primb, wtv, vf, NPRIMc, DPc, DPc, 0, 0, 0, 1, stream);
  run_convert(kf, kb, (long)NPRIMc * DPc, stream);
  {
    long per = (long)NPRIMc * DPc;
    dim3 g((unsigned)((per + 255) / 256), 1, 1);
    convert_t_bf16_kernel<<<g, 256, 0, stream>>>(vf, vtb, NPRIMc, DPc);
  }
  run_gemm(qtb, kb, scf, Nc, NPRIMc, DPc, 0, 0, 0, 1, stream);   // scores
  softmax512_kernel<<<Nc, 256, 0, stream>>>(scf, pb);
  run_gemm(pb, vtb, ctxf, Nc, DPc, NPRIMc, 0, 0, 0, 1, stream);  // P @ V
  run_convert(ctxf, ctxb, (long)Nc * DPc, stream);
  run_gemm(ctxb, wto, tprojf, Nc, Dc, DPc, 0, 0, 0, 1, stream);
  final_kernel<<<Nc, 256, 0, stream>>>(x2, tprojf, xn, tou_gate_w, tou_gate_b,
                                       out);
}
